// gruPredictor_22995254903302
// MI455X (gfx1250) — compile-verified, matrix operand order checked by reading, not run on hardware
//
#include <hip/hip_runtime.h>

// ---------------------------------------------------------------------------
// Problem constants (from reference)
// ---------------------------------------------------------------------------
#define TT 64
#define BB 256
#define DD 2000
#define TN 3000
#define EE 256
#define HH 256
#define AA 128
#define VV 128
#define PSZ 64
#define PEZ 64
#define SS 128
#define MROWS (TT * BB)   // 16384

typedef __bf16 bf16_t;
typedef __attribute__((ext_vector_type(16))) __bf16 v16bf;
typedef __attribute__((ext_vector_type(8)))  __bf16 v8bf;
typedef __attribute__((ext_vector_type(8)))  float  v8f;

// ---------------------------------------------------------------------------
// WMMA bf16 GEMM: C[M,N] = act(A[M,K] @ B + bias)
//   BT=false: B row-major [K,N], ldb = N
//   BT=true : B row-major [N,K], ldb = K   (multiply by B^T)
// Block tile 64(M) x 128(N), K-step 32, 8 waves, each wave 32x32 via 2x2
// v_wmma_f32_16x16x32_bf16 fragments. B is staged TRANSPOSED in LDS so both
// A and B fragments are contiguous 16B chunks (ds_load_b128). Full interior
// tiles take a vectorized float4 staging path; only edge tiles are guarded.
// ACT: 0=none 1=sigmoid 2=tanh 3=relu
// ---------------------------------------------------------------------------
#define BLK_M 64
#define BLK_N 128
#define BLK_K 32
#define LDK (BLK_K + 8)   // LDS row stride (bf16): 80 bytes, 16B aligned

template <int ACT, bool BT>
__global__ __launch_bounds__(256) void gemm_wmma_bf16(
    const float* __restrict__ A, int lda,
    const float* __restrict__ B, int ldb,
    float* __restrict__ C, int ldc,
    const float* __restrict__ bias,
    int M, int N, int K)
{
    __shared__ __align__(16) bf16_t lA[BLK_M][LDK];    // [m][k]
    __shared__ __align__(16) bf16_t lBt[BLK_N][LDK];   // [n][k] (transposed)

    const int tid  = threadIdx.x;
    const int lane = tid & 31;
    const int wave = tid >> 5;
    const int wm   = (wave & 1) * 32;
    const int wn   = (wave >> 1) * 32;
    const int blockM = blockIdx.y * BLK_M;
    const int blockN = blockIdx.x * BLK_N;

    // A staging coords: 8 consecutive K per thread
    const int ar  = tid >> 2;            // 0..63
    const int ac0 = (tid & 3) * 8;       // 0,8,16,24
    const int agm = blockM + ar;
    // B staging coords: 16 consecutive K per thread, one column
    const int bcol = tid & 127;          // 0..127
    const int bkh  = (tid >> 7) * 16;    // 0 or 16
    const int bgn  = blockN + bcol;

    v8f acc[2][2];
#pragma unroll
    for (int im = 0; im < 2; ++im)
#pragma unroll
        for (int in = 0; in < 2; ++in)
#pragma unroll
            for (int r = 0; r < 8; ++r) acc[im][in][r] = 0.0f;

    const int kTiles = (K + BLK_K - 1) / BLK_K;
    for (int kt = 0; kt < kTiles; ++kt) {
        const int k0 = kt * BLK_K;
        const bool fullK = (k0 + BLK_K <= K);

        // ---- stage A tile 64x32 ----
        if (agm < M && fullK) {
            const float4* src = (const float4*)(A + (size_t)agm * lda + k0 + ac0);
            const float4 x0 = src[0];
            const float4 x1 = src[1];
            v8bf d;
            d[0] = (bf16_t)x0.x; d[1] = (bf16_t)x0.y;
            d[2] = (bf16_t)x0.z; d[3] = (bf16_t)x0.w;
            d[4] = (bf16_t)x1.x; d[5] = (bf16_t)x1.y;
            d[6] = (bf16_t)x1.z; d[7] = (bf16_t)x1.w;
            *(v8bf*)&lA[ar][ac0] = d;
        } else {
            v8bf d;
#pragma unroll
            for (int i = 0; i < 8; ++i) d[i] = (bf16_t)0.0f;
            if (agm < M) {
#pragma unroll
                for (int i = 0; i < 8; ++i) {
                    const int gk = k0 + ac0 + i;
                    if (gk < K) d[i] = (bf16_t)A[(size_t)agm * lda + gk];
                }
            }
            *(v8bf*)&lA[ar][ac0] = d;
        }

        // ---- stage B tile 32x128 (transposed into lBt[n][k]) ----
        if (BT) {
            // B[N,K]: per-lane contiguous along K
            if (bgn < N && fullK) {
                const float4* src = (const float4*)(B + (size_t)bgn * ldb + k0 + bkh);
                const float4 x0 = src[0], x1 = src[1], x2 = src[2], x3 = src[3];
                v8bf d0, d1;
                d0[0] = (bf16_t)x0.x; d0[1] = (bf16_t)x0.y;
                d0[2] = (bf16_t)x0.z; d0[3] = (bf16_t)x0.w;
                d0[4] = (bf16_t)x1.x; d0[5] = (bf16_t)x1.y;
                d0[6] = (bf16_t)x1.z; d0[7] = (bf16_t)x1.w;
                d1[0] = (bf16_t)x2.x; d1[1] = (bf16_t)x2.y;
                d1[2] = (bf16_t)x2.z; d1[3] = (bf16_t)x2.w;
                d1[4] = (bf16_t)x3.x; d1[5] = (bf16_t)x3.y;
                d1[6] = (bf16_t)x3.z; d1[7] = (bf16_t)x3.w;
                *(v8bf*)&lBt[bcol][bkh]     = d0;
                *(v8bf*)&lBt[bcol][bkh + 8] = d1;
            } else {
                v8bf d0, d1;
#pragma unroll
                for (int i = 0; i < 8; ++i) { d0[i] = (bf16_t)0.0f; d1[i] = (bf16_t)0.0f; }
                if (bgn < N) {
#pragma unroll
                    for (int i = 0; i < 16; ++i) {
                        const int gk = k0 + bkh + i;
                        if (gk < K) {
                            const bf16_t v = (bf16_t)B[(size_t)bgn * ldb + gk];
                            if (i < 8) d0[i] = v; else d1[i - 8] = v;
                        }
                    }
                }
                *(v8bf*)&lBt[bcol][bkh]     = d0;
                *(v8bf*)&lBt[bcol][bkh + 8] = d1;
            }
        } else {
            // B[K,N]: strided along K, coalesced across lanes
            v8bf d0, d1;
#pragma unroll
            for (int i = 0; i < 8; ++i) { d0[i] = (bf16_t)0.0f; d1[i] = (bf16_t)0.0f; }
            if (bgn < N && fullK) {
#pragma unroll
                for (int i = 0; i < 16; ++i) {
                    const bf16_t v = (bf16_t)B[(size_t)(k0 + bkh + i) * ldb + bgn];
                    if (i < 8) d0[i] = v; else d1[i - 8] = v;
                }
            } else if (bgn < N) {
#pragma unroll
                for (int i = 0; i < 16; ++i) {
                    const int gk = k0 + bkh + i;
                    if (gk < K) {
                        const bf16_t v = (bf16_t)B[(size_t)gk * ldb + bgn];
                        if (i < 8) d0[i] = v; else d1[i - 8] = v;
                    }
                }
            }
            *(v8bf*)&lBt[bcol][bkh]     = d0;
            *(v8bf*)&lBt[bcol][bkh + 8] = d1;
        }
        __syncthreads();

        // ---- fragments (ISA lane map: lanes 0-15 K=[0..7]∪[16..23], 16-31 +8)
        const int half = lane & 15;
        const int kb   = (lane >> 4) * 8;

        v16bf afrag[2], bfrag[2];
#pragma unroll
        for (int im = 0; im < 2; ++im) {
            const bf16_t* pa = &lA[wm + im * 16 + half][0];
            const v8bf lo = *(const v8bf*)(pa + kb);
            const v8bf hi = *(const v8bf*)(pa + 16 + kb);
#pragma unroll
            for (int i = 0; i < 8; ++i) { afrag[im][i] = lo[i]; afrag[im][8 + i] = hi[i]; }
        }
#pragma unroll
        for (int in = 0; in < 2; ++in) {
            const bf16_t* pb = &lBt[wn + in * 16 + half][0];
            const v8bf lo = *(const v8bf*)(pb + kb);
            const v8bf hi = *(const v8bf*)(pb + 16 + kb);
#pragma unroll
            for (int i = 0; i < 8; ++i) { bfrag[in][i] = lo[i]; bfrag[in][8 + i] = hi[i]; }
        }

#pragma unroll
        for (int im = 0; im < 2; ++im)
#pragma unroll
            for (int in = 0; in < 2; ++in)
                acc[im][in] = __builtin_amdgcn_wmma_f32_16x16x32_bf16(
                    false, afrag[im], false, bfrag[in],
                    (short)0, acc[im][in], false, false);

        __syncthreads();
    }

    // ---- epilogue: C/D layout N = lane&15, M = vgpr + 8*(lane>=16) ----
    const int nCol = lane & 15;
    const int mHi  = (lane >> 4) * 8;
    const bool fullTile = (blockM + BLK_M <= M) && (blockN + BLK_N <= N);
#pragma unroll
    for (int im = 0; im < 2; ++im) {
#pragma unroll
        for (int in = 0; in < 2; ++in) {
            const int gn = blockN + wn + in * 16 + nCol;
            const float bv = (bias && gn < N) ? bias[gn] : 0.0f;
#pragma unroll
            for (int r = 0; r < 8; ++r) {
                const int gm = blockM + wm + im * 16 + mHi + r;
                float v = acc[im][in][r] + bv;
                if (ACT == 1)      v = 1.0f / (1.0f + __expf(-v));
                else if (ACT == 2) v = tanhf(v);
                else if (ACT == 3) v = v > 0.0f ? v : 0.0f;
                if (fullTile || (gm < M && gn < N))
                    C[(size_t)gm * ldc + gn] = v;
            }
        }
    }
}

// ---------------------------------------------------------------------------
// GRAM group attention: one block per (leaf,anc) group, 128 threads (= A)
// ---------------------------------------------------------------------------
__global__ __launch_bounds__(128) void group_embed_kernel(
    const float* __restrict__ embd, const float* __restrict__ Wa_w,
    const float* __restrict__ Wa_b, const float* __restrict__ Ua_w,
    const int* __restrict__ leaves, const int* __restrict__ anc,
    int L, float* __restrict__ out)
{
    const int n = blockIdx.x;
    const int tid = threadIdx.x;
    __shared__ float sred[6][128];
    __shared__ float swt[6];
    int ancIdx[6];

    for (int l = 0; l < L; ++l) {
        const int leaf = leaves[n * L + l];
        const int a    = anc[n * L + l];
        ancIdx[l] = a;
        const float* le = embd + (size_t)leaf * EE;
        const float* ae = embd + (size_t)a * EE;
        float accv = Wa_b[tid];
        for (int e = 0; e < EE; ++e) {
            accv += le[e] * Wa_w[(size_t)e * AA + tid];
            accv += ae[e] * Wa_w[(size_t)(EE + e) * AA + tid];
        }
        sred[l][tid] = tanhf(accv) * Ua_w[tid];
    }
    __syncthreads();
    for (int o = 64; o > 0; o >>= 1) {
        if (tid < o)
            for (int l = 0; l < L; ++l) sred[l][tid] += sred[l][tid + o];
        __syncthreads();
    }
    if (tid == 0) {
        float mx = sred[0][0];
        for (int l = 1; l < L; ++l) mx = fmaxf(mx, sred[l][0]);
        float sum = 0.0f;
        for (int l = 0; l < L; ++l) { float e = __expf(sred[l][0] - mx); swt[l] = e; sum += e; }
        const float inv = 1.0f / sum;
        for (int l = 0; l < L; ++l) swt[l] *= inv;
    }
    __syncthreads();
    for (int e = tid; e < EE; e += 128) {
        float o = 0.0f;
        for (int l = 0; l < L; ++l) o += swt[l] * embd[(size_t)ancIdx[l] * EE + e];
        out[(size_t)n * EE + e] = o;
    }
}

// ---------------------------------------------------------------------------
// small gathers / elementwise
// ---------------------------------------------------------------------------
__global__ void gather_rows_kernel(const float* __restrict__ src,
                                   const int* __restrict__ idx,
                                   float* __restrict__ dst, int rows, int cols)
{
    size_t i = (size_t)blockIdx.x * blockDim.x + threadIdx.x;
    if (i >= (size_t)rows * cols) return;
    int r = (int)(i / cols), c = (int)(i % cols);
    dst[i] = src[(size_t)idx[r] * cols + c];
}

__global__ void km_transpose_kernel(const float* __restrict__ embd,
                                    const int* __restrict__ ids,
                                    float* __restrict__ dst)  // [E,S]
{
    int i = blockIdx.x * blockDim.x + threadIdx.x;
    if (i >= EE * SS) return;
    int e = i / SS, s = i % SS;
    dst[i] = embd[(size_t)ids[s] * EE + e];
}

__global__ void prof_embed_kernel(const float* __restrict__ profiles,
                                  const float* __restrict__ w,
                                  const float* __restrict__ b,
                                  float* __restrict__ out)  // [B,PE]
{
    int i = blockIdx.x * blockDim.x + threadIdx.x;
    if (i >= BB * PEZ) return;
    int bb = i / PEZ, p = i % PEZ;
    float acc = b[p];
    for (int k = 0; k < PSZ; ++k) acc += profiles[bb * PSZ + k] * w[k * PEZ + p];
    out[i] = acc;
}

__global__ void gru_update_kernel(const float* __restrict__ Gx,  // [T,B,3H]
                                  const float* __restrict__ Gh,  // [B,3H]
                                  float* __restrict__ h,         // [B,H]
                                  float* __restrict__ gruOut,    // [T,B,H]
                                  const int* __restrict__ lens, int t)
{
    int i = blockIdx.x * blockDim.x + threadIdx.x;  // B*H
    if (i >= BB * HH) return;
    int b = i / HH, j = i % HH;
    size_t gxo = ((size_t)t * BB + b) * (3 * HH);
    size_t gho = (size_t)b * (3 * HH);
    float r = 1.0f / (1.0f + __expf(-(Gx[gxo + j] + Gh[gho + j])));
    float z = 1.0f / (1.0f + __expf(-(Gx[gxo + HH + j] + Gh[gho + HH + j])));
    float n = tanhf(Gx[gxo + 2 * HH + j] + r * Gh[gho + 2 * HH + j]);
    float hold = h[i];
    float hnew = (1.0f - z) * n + z * hold;
    bool m = t < lens[b];
    float hout = m ? hnew : hold;
    h[i] = hout;
    gruOut[((size_t)t * BB + b) * HH + j] = m ? hout : 0.0f;
}

__global__ void build_cat1_kernel(const float* __restrict__ gruOut,
                                  const float* __restrict__ prof,
                                  float* __restrict__ cat1)  // [16384, 320]
{
    size_t i = (size_t)blockIdx.x * blockDim.x + threadIdx.x;
    if (i >= (size_t)MROWS * (HH + PEZ)) return;
    int row = (int)(i / (HH + PEZ)), j = (int)(i % (HH + PEZ));
    int b = row % BB;
    cat1[i] = (j < HH) ? gruOut[(size_t)row * HH + j] : prof[b * PEZ + (j - HH)];
}

__global__ void build_catA_kernel(const float* __restrict__ prof,
                                  const float* __restrict__ readB,
                                  float* __restrict__ catA)  // [16384, 192]
{
    size_t i = (size_t)blockIdx.x * blockDim.x + threadIdx.x;
    if (i >= (size_t)MROWS * (PEZ + VV)) return;
    int row = (int)(i / (PEZ + VV)), j = (int)(i % (PEZ + VV));
    int b = row % BB;
    catA[i] = (j < PEZ) ? prof[b * PEZ + j] : readB[(size_t)row * VV + (j - PEZ)];
}

__global__ void build_cat2_kernel(const float* __restrict__ gruOut,
                                  const float* __restrict__ readB,
                                  const float* __restrict__ prof,
                                  const float* __restrict__ attn,
                                  float* __restrict__ cat2)  // [16384, 448]
{
    size_t i = (size_t)blockIdx.x * blockDim.x + threadIdx.x;
    const int W = HH + VV + PEZ;
    if (i >= (size_t)MROWS * W) return;
    int row = (int)(i / W), j = (int)(i % W);
    int b = row % BB;
    float v;
    if (j < HH)           v = gruOut[(size_t)row * HH + j];
    else if (j < HH + VV) v = readB[(size_t)row * VV + (j - HH)];
    else { int p = j - HH - VV; v = prof[b * PEZ + p] * attn[(size_t)row * PEZ + p]; }
    cat2[i] = v;
}

// ---------------------------------------------------------------------------
// DKVMN memory scan. One block per batch element, VM[b] (64 KB) stays hot in
// the 192 MB L2; t-loop inside the kernel so there is exactly one launch.
// ---------------------------------------------------------------------------
__global__ __launch_bounds__(128) void mem_scan_kernel(
    const float* __restrict__ slotw,   // [T,B,S]
    const float* __restrict__ er,      // [T,B,V]
    const float* __restrict__ ad,      // [T,B,V]
    const float* __restrict__ iniVam,  // [S,V]
    float* __restrict__ VM,            // [B,S,V]
    float* __restrict__ readOut)       // [T,B,V]
{
    const int b = blockIdx.x;
    const int v = threadIdx.x;  // V = 128
    __shared__ float sw[SS];
    float* vm = VM + (size_t)b * SS * VV;
    for (int s = 0; s < SS; ++s) vm[s * VV + v] = iniVam[s * VV + v];
    for (int t = 0; t < TT; ++t) {
        const size_t row = (size_t)t * BB + b;
        __syncthreads();
        sw[v] = slotw[row * SS + v];  // S == V == 128
        __syncthreads();
        const float e = er[row * VV + v];
        const float a = ad[row * VV + v];
        float acc = 0.0f;
        for (int s = 0; s < SS; ++s) {
            const float w = sw[s];
            const float m = vm[s * VV + v];
            acc += w * m;                                   // read (pre-update)
            vm[s * VV + v] = m * (1.0f - w * e) + w * a;    // erase/add
        }
        readOut[row * VV + v] = acc;
    }
}

// ---------------------------------------------------------------------------
// row softmax (in place)
// ---------------------------------------------------------------------------
__global__ __launch_bounds__(256) void softmax_rows_kernel(float* __restrict__ data, int cols)
{
    __shared__ float red[256];
    float* p = data + (size_t)blockIdx.x * cols;
    float mx = -3.402823e38f;
    for (int c = threadIdx.x; c < cols; c += blockDim.x) mx = fmaxf(mx, p[c]);
    red[threadIdx.x] = mx; __syncthreads();
    for (int o = 128; o > 0; o >>= 1) {
        if ((int)threadIdx.x < o) red[threadIdx.x] = fmaxf(red[threadIdx.x], red[threadIdx.x + o]);
        __syncthreads();
    }
    mx = red[0]; __syncthreads();
    float sum = 0.0f;
    for (int c = threadIdx.x; c < cols; c += blockDim.x) {
        float e = __expf(p[c] - mx); p[c] = e; sum += e;
    }
    red[threadIdx.x] = sum; __syncthreads();
    for (int o = 128; o > 0; o >>= 1) {
        if ((int)threadIdx.x < o) red[threadIdx.x] += red[threadIdx.x + o];
        __syncthreads();
    }
    const float inv = 1.0f / red[0];
    for (int c = threadIdx.x; c < cols; c += blockDim.x) p[c] *= inv;
}

// ---------------------------------------------------------------------------
// launcher
// ---------------------------------------------------------------------------
extern "C" void kernel_launch(void* const* d_in, const int* in_sizes, int n_in,
                              void* d_out, int out_size, void* d_ws, size_t ws_size,
                              hipStream_t stream)
{
    (void)in_sizes; (void)n_in; (void)out_size; (void)ws_size;
    const float* X         = (const float*)d_in[0];
    const float* profiles  = (const float*)d_in[1];
    const float* ini_embd  = (const float*)d_in[2];
    const float* Wa_w      = (const float*)d_in[3];
    const float* Wa_b      = (const float*)d_in[4];
    const float* Ua_w      = (const float*)d_in[5];
    const float* gru_Wih   = (const float*)d_in[6];
    const float* gru_Whh   = (const float*)d_in[7];
    const float* gru_bih   = (const float*)d_in[8];
    const float* gru_bhh   = (const float*)d_in[9];
    const float* tranH_w   = (const float*)d_in[10];
    const float* tranH_b   = (const float*)d_in[11];
    const float* out_w     = (const float*)d_in[12];
    const float* out_b     = (const float*)d_in[13];
    const float* erase_w   = (const float*)d_in[14];
    const float* erase_b   = (const float*)d_in[15];
    const float* add_w     = (const float*)d_in[16];
    const float* add_b     = (const float*)d_in[17];
    const float* iniVam    = (const float*)d_in[18];
    const float* emP_w     = (const float*)d_in[19];
    const float* emP_b     = (const float*)d_in[20];
    const float* attnP_w   = (const float*)d_in[21];
    const float* attnP_b   = (const float*)d_in[22];
    const int*   KMIds     = (const int*)d_in[23];
    const int*   leaves1   = (const int*)d_in[24];
    const int*   anc1      = (const int*)d_in[25];
    const int*   leaves2   = (const int*)d_in[26];
    const int*   anc2      = (const int*)d_in[27];
    const int*   mapInfo   = (const int*)d_in[28];
    const int*   X_lengths = (const int*)d_in[29];

    // workspace layout (float offsets)
    float* ws = (float*)d_ws;
    float* embedTmp = ws;                               // 3000*256
    float* embedMat = embedTmp + (size_t)TN * EE;       // 2000*256
    float* Xe       = embedMat + (size_t)DD * EE;       // 16384*256
    float* Gx       = Xe + (size_t)MROWS * EE;          // 16384*768 (reused later)
    float* h        = Gx + (size_t)MROWS * 3 * HH;      // 256*256
    float* Gh       = h + (size_t)BB * HH;              // 256*768
    float* gruOut   = Gh + (size_t)BB * 3 * HH;         // 16384*256
    float* prof     = gruOut + (size_t)MROWS * HH;      // 256*64
    float* cat1     = prof + (size_t)BB * PEZ;          // 16384*320
    float* tran     = cat1 + (size_t)MROWS * (HH+PEZ);  // 16384*256
    float* KMT      = tran + (size_t)MROWS * EE;        // 256*128
    float* slotw    = KMT + (size_t)EE * SS;            // 16384*128
    float* eraseB   = slotw + (size_t)MROWS * SS;       // 16384*128
    float* addB     = eraseB + (size_t)MROWS * VV;      // 16384*128
    float* readB    = addB + (size_t)MROWS * VV;        // 16384*128
    // Gx region (12.58M floats) reused after the GRU loop:
    float* VM      = Gx;                                //  4.19M (scan only)
    float* catA    = Gx;                                //  3.15M (after scan)
    float* attnBuf = Gx + (size_t)MROWS * (PEZ + VV);   //  1.05M
    float* cat2    = Gx + (size_t)BB * SS * VV;         //  7.34M (no overlap w/ attnBuf)

    float* out = (float*)d_out;  // [16384, 2000] logits -> softmax in place

    // 1) GRAM group attention -> embedTmp[3000,256]
    group_embed_kernel<<<1500, 128, 0, stream>>>(ini_embd, Wa_w, Wa_b, Ua_w,
                                                 leaves1, anc1, 4, embedTmp);
    group_embed_kernel<<<1500, 128, 0, stream>>>(ini_embd, Wa_w, Wa_b, Ua_w,
                                                 leaves2, anc2, 6,
                                                 embedTmp + (size_t)1500 * EE);
    // 2) embedMat = embedTmp[mapInfo]  [D,E]
    gather_rows_kernel<<<(DD * EE + 255) / 256, 256, 0, stream>>>(
        embedTmp, mapInfo, embedMat, DD, EE);

    // 3) Xe = X @ embedMat  (16384 x 2000 x 256)
    gemm_wmma_bf16<0, false><<<dim3(2, 256), 256, 0, stream>>>(
        X, DD, embedMat, EE, Xe, EE, nullptr, MROWS, EE, DD);

    // 4) Gx = Xe @ Wih^T + bih  (16384 x 256 x 768), hoisted out of the scan
    gemm_wmma_bf16<0, true><<<dim3(6, 256), 256, 0, stream>>>(
        Xe, EE, gru_Wih, EE, Gx, 3 * HH, gru_bih, MROWS, 3 * HH, EE);

    // 5) GRU recurrence
    hipMemsetAsync(h, 0, (size_t)BB * HH * sizeof(float), stream);
    for (int t = 0; t < TT; ++t) {
        gemm_wmma_bf16<0, true><<<dim3(6, 4), 256, 0, stream>>>(
            h, HH, gru_Whh, HH, Gh, 3 * HH, gru_bhh, BB, 3 * HH, HH);
        gru_update_kernel<<<(BB * HH) / 256, 256, 0, stream>>>(
            Gx, Gh, h, gruOut, X_lengths, t);
    }

    // 6) profile embedding [B,PE]
    prof_embed_kernel<<<(BB * PEZ + 255) / 256, 256, 0, stream>>>(
        profiles, emP_w, emP_b, prof);

    // 7) tran = [gruOut | prof] @ tranH_w + b  (16384 x 320 x 256)
    build_cat1_kernel<<<((size_t)MROWS * (HH + PEZ) + 255) / 256, 256, 0, stream>>>(
        gruOut, prof, cat1);
    gemm_wmma_bf16<0, false><<<dim3(2, 256), 256, 0, stream>>>(
        cat1, HH + PEZ, tranH_w, EE, tran, EE, tranH_b, MROWS, EE, HH + PEZ);

    // 8) slot_w = softmax(tran @ KM^T)
    km_transpose_kernel<<<(EE * SS + 255) / 256, 256, 0, stream>>>(ini_embd, KMIds, KMT);
    gemm_wmma_bf16<0, false><<<dim3(1, 256), 256, 0, stream>>>(
        tran, EE, KMT, SS, slotw, SS, nullptr, MROWS, SS, EE);
    softmax_rows_kernel<<<MROWS, 256, 0, stream>>>(slotw, SS);

    // 9) erase = sigmoid(Xe@We+b), add = tanh(Xe@Wa+b)
    gemm_wmma_bf16<1, false><<<dim3(1, 256), 256, 0, stream>>>(
        Xe, EE, erase_w, VV, eraseB, VV, erase_b, MROWS, VV, EE);
    gemm_wmma_bf16<2, false><<<dim3(1, 256), 256, 0, stream>>>(
        Xe, EE, add_w, VV, addB, VV, add_b, MROWS, VV, EE);

    // 10) memory scan (VM stays in L2; Gx region is dead and hosts VM)
    mem_scan_kernel<<<BB, 128, 0, stream>>>(slotw, eraseB, addB, iniVam, VM, readB);

    // 11) attn = relu([prof | read] @ attnP_w + b)
    build_catA_kernel<<<((size_t)MROWS * (PEZ + VV) + 255) / 256, 256, 0, stream>>>(
        prof, readB, catA);
    gemm_wmma_bf16<3, false><<<dim3(1, 256), 256, 0, stream>>>(
        catA, PEZ + VV, attnP_w, PEZ, attnBuf, PEZ, attnP_b, MROWS, PEZ, PEZ + VV);

    // 12) final = [gruOut | read | prof*attn] @ out_w + b, softmax over D
    build_cat2_kernel<<<((size_t)MROWS * (HH + VV + PEZ) + 255) / 256, 256, 0, stream>>>(
        gruOut, readB, prof, attnBuf, cat2);
    gemm_wmma_bf16<0, false><<<dim3(16, 256), 256, 0, stream>>>(
        cat2, HH + VV + PEZ, out_w, DD, out, DD, out_b, MROWS, DD, HH + VV + PEZ);
    softmax_rows_kernel<<<MROWS, 256, 0, stream>>>(out, DD);
}
